// MultiHeadedLinrec_30399778521713
// MI455X (gfx1250) — matrix operand order repaired and checked by reading, not compiled, not run on hardware
//
#include <hip/hip_runtime.h>
#include <hip/hip_bf16.h>

typedef __bf16 bf16;
typedef __bf16 v16bf __attribute__((ext_vector_type(16)));
typedef __bf16 v8bf  __attribute__((ext_vector_type(8)));
typedef float  v8f   __attribute__((ext_vector_type(8)));

#define HEADS 16
#define EMB   1024
#define BB    8
#define SS    4096
#define DH    64            // EMB / HEADS
#define MTOT  (BB * SS)     // 32768

// Build a 16x32 (A) or 32x16 (B) bf16 WMMA fragment from two 16-byte LDS chunks.
__device__ __forceinline__ v16bf ld2(const bf16* p0, const bf16* p1) {
    union { v16bf v; v8bf h[2]; } u;
    u.h[0] = *(const v8bf*)p0;
    u.h[1] = *(const v8bf*)p1;
    return u.v;
}

__device__ __forceinline__ v8f wmma_bf16(v16bf a, v16bf b, v8f c) {
    return __builtin_amdgcn_wmma_f32_16x16x32_bf16(false, a, false, b, (short)0, c, false, false);
}

// 64 contiguous bytes global -> LDS per lane via gfx1250 async-tensor path.
// INST_OFFSET is applied to both the global and the LDS address, so four
// b128 copies at offsets 0/16/32/48 move the whole chunk. Tracked by ASYNCcnt.
__device__ __forceinline__ void async_copy64(const bf16* g, bf16* s) {
    unsigned lds = (unsigned)(uintptr_t)s;                 // low 32 bits = LDS offset
    unsigned long long ga = (unsigned long long)(uintptr_t)g;
    asm volatile(
        "global_load_async_to_lds_b128 %0, %1, off\n\t"
        "global_load_async_to_lds_b128 %0, %1, off offset:16\n\t"
        "global_load_async_to_lds_b128 %0, %1, off offset:32\n\t"
        "global_load_async_to_lds_b128 %0, %1, off offset:48"
        :: "v"(lds), "v"(ga) : "memory");
}

__device__ __forceinline__ void wait_async0() {
    asm volatile("s_wait_asynccnt 0x0" ::: "memory");
}

// ---------------------------------------------------------------------------
// f32 -> bf16 convert (4 elements / thread; all sizes are multiples of 4)
// ---------------------------------------------------------------------------
__global__ void cvt_kernel(const float* __restrict__ src, bf16* __restrict__ dst,
                           long long n) {
    long long i = ((long long)blockIdx.x * 256 + threadIdx.x) * 4;
    if (i < n) {
        float4 f = *(const float4*)(src + i);
        dst[i + 0] = (bf16)f.x;
        dst[i + 1] = (bf16)f.y;
        dst[i + 2] = (bf16)f.z;
        dst[i + 3] = (bf16)f.w;
    }
}

// ---------------------------------------------------------------------------
// Y = act(X @ W^T + b)   X:[M,K] bf16, W:[N,K] bf16, b:[N] f32
// BM=128 BN=128 BK=64, 256 threads = 8 waves in a 4(m) x 2(n) grid,
// wave tile 32x64 = 2x4 WMMA tiles. Double-buffered LDS filled with
// global_load_async_to_lds_b128, software-pipelined one K-slab ahead.
// ---------------------------------------------------------------------------
template<bool ELU, bool OUTF32>
__global__ __launch_bounds__(256)
void gemm_bf16(const bf16* __restrict__ X, const bf16* __restrict__ W,
               const float* __restrict__ bias, void* __restrict__ Yp,
               int M, int N, int K) {
    const int LDA = 72, LDB = 72;              // bf16 units; 144B rows (16B aligned)
    __shared__ __align__(16) bf16 As[2][128 * 72];
    __shared__ __align__(16) bf16 Bs[2][128 * 72];

    const int tid  = threadIdx.x;
    const int lane = tid & 31;
    const int wid  = tid >> 5;
    const int wm   = wid >> 1;                 // 0..3
    const int wn   = wid & 1;                  // 0..1
    const int m0   = blockIdx.y * 128;
    const int n0   = blockIdx.x * 128;

    v8f c[2][4] = {};

    const int lrow  = tid >> 1;                // 0..127
    const int lhalf = (tid & 1) * 32;          // 0 or 32 (bf16 elems)
    const int soff  = lrow * LDA + lhalf;      // same for A and B staging

    const bf16* gA = X + (size_t)(m0 + lrow) * K + lhalf;
    const bf16* gB = W + (size_t)(n0 + lrow) * K + lhalf;

    // prologue: fill buffer 0 with K-slab 0
    async_copy64(gA, &As[0][soff]);
    async_copy64(gB, &Bs[0][soff]);

    int buf = 0;
    for (int k0 = 0; k0 < K; k0 += 64, buf ^= 1) {
        wait_async0();                         // this wave's slab landed in LDS
        __syncthreads();                       // everyone's landed; prev compute done
        if (k0 + 64 < K) {                     // prefetch next slab into other buffer
            async_copy64(gA + k0 + 64, &As[buf ^ 1][soff]);
            async_copy64(gB + k0 + 64, &Bs[buf ^ 1][soff]);
        }

#pragma unroll
        for (int kk = 0; kk < 64; kk += 32) {
            const int ca = kk + ((lane < 16) ? 0 : 16);   // A frag K-chunk origin
            const int cb = kk + ((lane < 16) ? 0 : 16);   // B frag K-chunk origin
            v16bf afrag[2], bfrag[4];
#pragma unroll
            for (int i = 0; i < 2; i++) {
                const bf16* p = &As[buf][(wm * 32 + i * 16 + (lane & 15)) * LDA + ca];
                afrag[i] = ld2(p, p + 8);
            }
#pragma unroll
            for (int j = 0; j < 4; j++) {
                const bf16* p = &Bs[buf][(wn * 64 + j * 16 + (lane & 15)) * LDB + cb];
                bfrag[j] = ld2(p, p + 8);
            }
#pragma unroll
            for (int i = 0; i < 2; i++)
#pragma unroll
                for (int j = 0; j < 4; j++)
                    c[i][j] = wmma_bf16(afrag[i], bfrag[j], c[i][j]);
        }
    }

    // epilogue: C layout -> row = base + v + (lane<16?0:8), col = lane&15
    const int rbase = (lane < 16) ? 0 : 8;
    const int col15 = lane & 15;
#pragma unroll
    for (int j = 0; j < 4; j++) {
        const int col = n0 + wn * 64 + j * 16 + col15;
        const float bj = bias[col];
#pragma unroll
        for (int i = 0; i < 2; i++) {
#pragma unroll
            for (int v = 0; v < 8; v++) {
                const int row = m0 + wm * 32 + i * 16 + rbase + v;
                float x = c[i][j][v] + bj;
                if (ELU) x = (x > 0.f) ? x : (__expf(x) - 1.f);
                if (OUTF32) ((float*)Yp)[(size_t)row * N + col] = x;
                else        ((bf16*)Yp)[(size_t)row * N + col]  = (bf16)x;
            }
        }
    }
}

// ---------------------------------------------------------------------------
// Deterministic partial sum-of-squares of k over S (8 S-chunks, no atomics).
// part[(sc*BB + b)*EMB + n] = sum_{s in chunk} K[b,s,n]^2
// ---------------------------------------------------------------------------
__global__ __launch_bounds__(256)
void ksq_partial(const bf16* __restrict__ Kp, float* __restrict__ part) {
    const int b  = blockIdx.x;
    const int n  = blockIdx.y * 256 + threadIdx.x;
    const int sc = blockIdx.z;
    const bf16* base = Kp + ((size_t)b * SS + sc * 512) * EMB + n;
    float acc = 0.f;
    for (int s = 0; s < 512; s++) {
        float x = (float)base[(size_t)s * EMB];
        acc = fmaf(x, x, acc);
    }
    part[((size_t)sc * BB + b) * EMB + n] = acc;
}

// ---------------------------------------------------------------------------
// scoresT[b,h,e,d] = (1/(||k_d|| * sqrt(S))) * sum_s V[b,s,h,e] * K[b,s,h,d]
// One block per (b,h). S-contraction via WMMA with transposed LDS staging.
// Waves: 4(m=e) x 2(n-halves); wave tile 16 x 32.
// ---------------------------------------------------------------------------
__global__ __launch_bounds__(256)
void scores_kernel(const bf16* __restrict__ Kp, const bf16* __restrict__ Vp,
                   const float* __restrict__ ksqpart, bf16* __restrict__ scoresT) {
    const int LDT = 136;                       // 272B rows, 16B aligned
    __shared__ __align__(16) bf16 Vt[64 * 136];   // [e][s]
    __shared__ __align__(16) bf16 Kt[64 * 136];   // [d][s]

    const int bh = blockIdx.x;
    const int b = bh / HEADS, h = bh % HEADS;
    const int tid = threadIdx.x, lane = tid & 31, wid = tid >> 5;
    const int wm = wid & 3, wn = wid >> 2;

    v8f c[2] = {};

    const bf16* Kb = Kp + (size_t)b * SS * EMB + h * DH;
    const bf16* Vb = Vp + (size_t)b * SS * EMB + h * DH;

    for (int s0 = 0; s0 < SS; s0 += 128) {
        __syncthreads();
        for (int idx = tid; idx < 128 * 8; idx += 256) {
            const int srow = idx >> 3;
            const int grp  = (idx & 7) * 8;
            union { uint4 u; bf16 e[8]; } kv, vv;
            kv.u = *(const uint4*)(Kb + (size_t)(s0 + srow) * EMB + grp);
            vv.u = *(const uint4*)(Vb + (size_t)(s0 + srow) * EMB + grp);
#pragma unroll
            for (int q = 0; q < 8; q++) {
                Kt[(grp + q) * LDT + srow] = kv.e[q];
                Vt[(grp + q) * LDT + srow] = vv.e[q];
            }
        }
        __syncthreads();
#pragma unroll
        for (int kk = 0; kk < 128; kk += 32) {
            const int ca = kk + ((lane < 16) ? 0 : 16);
            const int cb = kk + ((lane < 16) ? 0 : 16);
            const bf16* pa = Vt + (wm * 16 + (lane & 15)) * LDT + ca;
            v16bf a = ld2(pa, pa + 8);
#pragma unroll
            for (int j = 0; j < 2; j++) {
                const bf16* pb = Kt + (wn * 32 + j * 16 + (lane & 15)) * LDT + cb;
                c[j] = wmma_bf16(a, ld2(pb, pb + 8), c[j]);
            }
        }
    }

    const int rbase = (lane < 16) ? 0 : 8;
    const int col15 = lane & 15;
    bf16* out = scoresT + (size_t)bh * DH * DH;
#pragma unroll
    for (int j = 0; j < 2; j++) {
        const int d = wn * 32 + j * 16 + col15;
        float ssq = 0.f;
#pragma unroll
        for (int sc = 0; sc < 8; sc++)
            ssq += ksqpart[((size_t)sc * BB + b) * EMB + h * DH + d];
        const float inv = __frsqrt_rn(ssq) * (1.0f / 64.0f);   // 1/(||k||*sqrt(S))
#pragma unroll
        for (int v = 0; v < 8; v++) {
            const int e = wm * 16 + rbase + v;
            out[e * DH + d] = (bf16)(c[j][v] * inv);
        }
    }
}

// ---------------------------------------------------------------------------
// x[b,s,h,e] = (1/(||q_row||*sqrt(d))) * sum_d q[b,s,h,d] * scoresT[b,h,e,d]
// One block per (b,h,s-tile of 128). Wave w owns m-tile w (16 rows) x full N=64.
// ---------------------------------------------------------------------------
__global__ __launch_bounds__(256)
void attn_kernel(const bf16* __restrict__ Qp, const bf16* __restrict__ scoresT,
                 bf16* __restrict__ Xo) {
    const int LDA = 72, LDB = 72;
    __shared__ __align__(16) bf16  Qs[128 * 72];
    __shared__ __align__(16) bf16  Bs[64 * 72];
    __shared__ float red[256];
    __shared__ float invn[128];

    const int bh = blockIdx.y;
    const int b = bh / HEADS, h = bh % HEADS;
    const int s0 = blockIdx.x * 128;
    const int tid = threadIdx.x, lane = tid & 31, wid = tid >> 5;

    const int lrow  = tid >> 1;
    const int lhalf = (tid & 1) * 32;
    const bf16* Qb = Qp + ((size_t)b * SS + s0 + lrow) * EMB + h * DH + lhalf;

    float ssq = 0.f;
#pragma unroll
    for (int t = 0; t < 4; t++) {
        union { uint4 u; bf16 e[8]; } q;
        q.u = *(const uint4*)(Qb + t * 8);
#pragma unroll
        for (int i = 0; i < 8; i++) {
            float x = (float)q.e[i];
            ssq = fmaf(x, x, ssq);
        }
        *(uint4*)(Qs + lrow * LDA + lhalf + t * 8) = q.u;
    }
    red[tid] = ssq;

    const bf16* Sb = scoresT + (size_t)bh * DH * DH;
    for (int idx = tid; idx < 64 * 8; idx += 256) {
        const int er = idx >> 3;
        const int g  = (idx & 7) * 8;
        *(uint4*)(Bs + er * LDB + g) = *(const uint4*)(Sb + er * DH + g);
    }
    __syncthreads();
    if (tid < 128)
        invn[tid] = __frsqrt_rn(red[2 * tid] + red[2 * tid + 1]) * 0.125f; // 1/(||q||*8)
    __syncthreads();

    v8f c[4] = {};
#pragma unroll
    for (int kk = 0; kk < 64; kk += 32) {
        const int ca = kk + ((lane < 16) ? 0 : 16);
        const int cb = kk + ((lane < 16) ? 0 : 16);
        const bf16* pa = Qs + (wid * 16 + (lane & 15)) * LDA + ca;
        v16bf a = ld2(pa, pa + 8);
#pragma unroll
        for (int j = 0; j < 4; j++) {
            const bf16* pb = Bs + (j * 16 + (lane & 15)) * LDB + cb;
            c[j] = wmma_bf16(a, ld2(pb, pb + 8), c[j]);
        }
    }

    const int rbase = (lane < 16) ? 0 : 8;
    const int col15 = lane & 15;
#pragma unroll
    for (int j = 0; j < 4; j++) {
        const int col = j * 16 + col15;
#pragma unroll
        for (int v = 0; v < 8; v++) {
            const int row = wid * 16 + rbase + v;
            const float x = c[j][v] * invn[row];
            Xo[((size_t)b * SS + s0 + row) * EMB + h * DH + col] = (bf16)x;
        }
    }
}

// ---------------------------------------------------------------------------
extern "C" void kernel_launch(void* const* d_in, const int* in_sizes, int n_in,
                              void* d_out, int out_size, void* d_ws, size_t ws_size,
                              hipStream_t stream) {
    const float* query = (const float*)d_in[0];
    const float* key   = (const float*)d_in[1];
    const float* value = (const float*)d_in[2];
    const float* Wq = (const float*)d_in[3];
    const float* bq = (const float*)d_in[4];
    const float* Wk = (const float*)d_in[5];
    const float* bk = (const float*)d_in[6];
    const float* Wv = (const float*)d_in[7];
    const float* bv = (const float*)d_in[8];
    const float* Wo = (const float*)d_in[9];
    const float* bo = (const float*)d_in[10];

    const size_t nX = (size_t)BB * SS * EMB;   // 33,554,432
    const size_t nW = (size_t)EMB * EMB;       // 1,048,576

    char* ws = (char*)d_ws;
    bf16* qbf = (bf16*)ws; ws += nX * 2;
    bf16* kbf = (bf16*)ws; ws += nX * 2;
    bf16* vbf = (bf16*)ws; ws += nX * 2;
    bf16* wqb = (bf16*)ws; ws += nW * 2;
    bf16* wkb = (bf16*)ws; ws += nW * 2;
    bf16* wvb = (bf16*)ws; ws += nW * 2;
    bf16* wob = (bf16*)ws; ws += nW * 2;
    bf16* Qp  = (bf16*)ws; ws += nX * 2;
    bf16* Kp  = (bf16*)ws; ws += nX * 2;
    bf16* Vp  = (bf16*)ws; ws += nX * 2;
    float* ksqp = (float*)ws; ws += (size_t)8 * BB * EMB * 4;
    bf16* scT = (bf16*)ws; ws += (size_t)BB * HEADS * DH * DH * 2;
    bf16* Xat = (bf16*)ws; ws += nX * 2;

    // 1. convert inputs + weights to bf16
    cvt_kernel<<<(int)(nX / 4 / 256), 256, 0, stream>>>(query, qbf, (long long)nX);
    cvt_kernel<<<(int)(nX / 4 / 256), 256, 0, stream>>>(key,   kbf, (long long)nX);
    cvt_kernel<<<(int)(nX / 4 / 256), 256, 0, stream>>>(value, vbf, (long long)nX);
    cvt_kernel<<<(int)(nW / 4 / 256), 256, 0, stream>>>(Wq, wqb, (long long)nW);
    cvt_kernel<<<(int)(nW / 4 / 256), 256, 0, stream>>>(Wk, wkb, (long long)nW);
    cvt_kernel<<<(int)(nW / 4 / 256), 256, 0, stream>>>(Wv, wvb, (long long)nW);
    cvt_kernel<<<(int)(nW / 4 / 256), 256, 0, stream>>>(Wo, wob, (long long)nW);

    // 2. projections: q = elu(x Wq^T + bq), k = elu(...), v = (...)
    dim3 gg(EMB / 128, MTOT / 128);            // (8, 256)
    gemm_bf16<true,  false><<<gg, 256, 0, stream>>>(qbf, wqb, bq, (void*)Qp, MTOT, EMB, EMB);
    gemm_bf16<true,  false><<<gg, 256, 0, stream>>>(kbf, wkb, bk, (void*)Kp, MTOT, EMB, EMB);
    gemm_bf16<false, false><<<gg, 256, 0, stream>>>(vbf, wvb, bv, (void*)Vp, MTOT, EMB, EMB);

    // 3. k column sum-of-squares partials (deterministic, no atomics)
    dim3 gk(BB, EMB / 256, 8);
    ksq_partial<<<gk, 256, 0, stream>>>(Kp, ksqp);

    // 4. scoresT = V^T K with fused k-normalization, stored [b,h,e,d] bf16
    scores_kernel<<<BB * HEADS, 256, 0, stream>>>(Kp, Vp, ksqp, scT);

    // 5. x = qnorm @ scores -> [B,S,E] bf16
    dim3 ga(SS / 128, BB * HEADS);
    attn_kernel<<<ga, 256, 0, stream>>>(Qp, scT, Xat);

    // 6. out = x @ Wo^T + bo (f32)
    gemm_bf16<false, true><<<gg, 256, 0, stream>>>(Xat, wob, bo, d_out, MTOT, EMB, EMB);
}
